// GroupedQueryAttention_11742440587389
// MI455X (gfx1250) — matrix operand order repaired and checked by reading, not compiled, not run on hardware
//
#include <hip/hip_runtime.h>
#include <hip/hip_fp16.h>

typedef __attribute__((ext_vector_type(16))) _Float16 v16h;
typedef __attribute__((ext_vector_type(8)))  float    v8f;

constexpr int Bq   = 2;
constexpr int Sq   = 2048;
constexpr int Eq   = 2048;
constexpr int Hq   = 32;
constexpr int KVHq = 8;
constexpr int HDq  = 64;
constexpr float NEG_BIG = -3.0e38f;

union HFrag { v16h v; uint4 q[2]; };
union H8    { uint4 q; _Float16 h[8]; };

__device__ __forceinline__ v8f wmma16(const v16h& a, const v16h& b, const v8f& c) {
  // D(16x16 f32) = A(16x32 f16) * B(32x16 f16) + C
  return __builtin_amdgcn_wmma_f32_16x16x32_f16(false, a, false, b, (short)0, c,
                                                false, false);
}

__global__ void cvt_f32_f16(const float* __restrict__ in, _Float16* __restrict__ out,
                            int n) {
  int i = blockIdx.x * blockDim.x + threadIdx.x;
  if (i < n) out[i] = (_Float16)in[i];
}

// C[m,n] = sum_k A[m,k] * W[n,k] + bias[n]
// MODE 0: f32 output, row-major MxN (final projection -> d_out)
// MODE 1: f16 output scattered to per-head layout ((b*NH + n/64)*S + s)*64 + n%64
// 128x128 block tile, 8 waves of 32x64, K-chunk 64 with register double-buffering.
template <int MODE>
__global__ __launch_bounds__(256) void gemm_f16_wmma(
    const _Float16* __restrict__ A,   // M x K
    const _Float16* __restrict__ W,   // N x K
    const float* __restrict__ bias,   // N
    void* __restrict__ outp,
    int M, int N, int K, int NH)
{
  constexpr int LDT = 72;                 // padded row stride (halves), 144B: 16B aligned
  __shared__ _Float16 lA[128 * LDT];
  __shared__ _Float16 lB[128 * LDT];

  const int tid  = threadIdx.x;
  const int lane = tid & 31;
  const int wave = tid >> 5;
  const int wm   = wave & 3;              // wave M position (4 x 32 rows)
  const int wn   = wave >> 2;             // wave N position (2 x 64 cols)
  const int lr   = lane & 15;
  const int hl   = lane >> 4;

  const int m0 = blockIdx.y * 128;
  const int n0 = blockIdx.x * 128;

  const int row0 = tid >> 3;              // staging: 128 rows x 8 segs of 16B
  const int seg  = tid & 7;

  v8f acc[2][4];
  #pragma unroll
  for (int mf = 0; mf < 2; ++mf)
    #pragma unroll
    for (int nf = 0; nf < 4; ++nf) acc[mf][nf] = {};

  // preload K-chunk 0 into registers
  uint4 pa[4], pb[4];
  #pragma unroll
  for (int j = 0; j < 4; ++j) {
    int row = row0 + 32 * j;
    pa[j] = *(const uint4*)&A[(size_t)(m0 + row) * K + seg * 8];
    pb[j] = *(const uint4*)&W[(size_t)(n0 + row) * K + seg * 8];
  }

  for (int kc = 0; kc < K; kc += 64) {
    __syncthreads();
    #pragma unroll
    for (int j = 0; j < 4; ++j) {         // VGPR -> LDS (waits on the global loads)
      int row = row0 + 32 * j;
      *(uint4*)&lA[row * LDT + seg * 8] = pa[j];
      *(uint4*)&lB[row * LDT + seg * 8] = pb[j];
    }
    __syncthreads();

    if (kc + 64 < K) {                    // issue next chunk's global loads now;
      #pragma unroll                      // latency hidden behind 16 WMMAs below
      for (int j = 0; j < 4; ++j) {
        int row = row0 + 32 * j;
        pa[j] = *(const uint4*)&A[(size_t)(m0 + row) * K + kc + 64 + seg * 8];
        pb[j] = *(const uint4*)&W[(size_t)(n0 + row) * K + kc + 64 + seg * 8];
      }
      if (kc + 128 < K)                   // L2 prefetch one more chunk ahead
        __builtin_prefetch(&A[(size_t)(m0 + row0) * K + kc + 128 + seg * 8], 0, 1);
    }

    #pragma unroll
    for (int ks = 0; ks < 64; ks += 32) {
      HFrag a[2], b[4];
      #pragma unroll
      for (int mf = 0; mf < 2; ++mf) {    // A-layout: lane-half K offsets {0,8}+{16,24}
        int row = wm * 32 + mf * 16 + lr;
        a[mf].q[0] = *(const uint4*)&lA[row * LDT + ks + hl * 8];
        a[mf].q[1] = *(const uint4*)&lA[row * LDT + ks + 16 + hl * 8];
      }
      #pragma unroll
      for (int nf = 0; nf < 4; ++nf) {    // B-layout: lane-half K offsets {0,8}+{16,24}
        int row = wn * 64 + nf * 16 + lr;
        b[nf].q[0] = *(const uint4*)&lB[row * LDT + ks + hl * 8];
        b[nf].q[1] = *(const uint4*)&lB[row * LDT + ks + 16 + hl * 8];
      }
      #pragma unroll
      for (int mf = 0; mf < 2; ++mf)
        #pragma unroll
        for (int nf = 0; nf < 4; ++nf)
          acc[mf][nf] = wmma16(a[mf].v, b[nf].v, acc[mf][nf]);
    }
  }

  #pragma unroll
  for (int mf = 0; mf < 2; ++mf) {
    #pragma unroll
    for (int nf = 0; nf < 4; ++nf) {
      #pragma unroll
      for (int r = 0; r < 8; ++r) {       // C-layout: row = r + 8*hl, col = lr
        int gm = m0 + wm * 32 + mf * 16 + r + 8 * hl;
        int gn = n0 + wn * 64 + nf * 16 + lr;
        float v = acc[mf][nf][r] + bias[gn];
        if (MODE == 0) {
          ((float*)outp)[(size_t)gm * N + gn] = v;
        } else {
          int bb = gm / Sq, ss = gm % Sq;
          int hh = gn >> 6, hd = gn & 63;
          ((_Float16*)outp)[(((size_t)(bb * NH + hh)) * Sq + ss) * HDq + hd] =
              (_Float16)v;
        }
      }
    }
  }
}

// Flash attention: block = (b,h) x 128 query rows; wave = 16 query rows.
__global__ __launch_bounds__(256) void attn_flash_wmma(
    const _Float16* __restrict__ qh,   // (B*H,   S, HD)
    const _Float16* __restrict__ kh,   // (B*KVH, S, HD)
    const _Float16* __restrict__ vh,   // (B*KVH, S, HD)
    _Float16* __restrict__ ctxh)       // (B, S, E)
{
  constexpr int LDK = 72;              // padded stride (halves), 144B: 16B aligned
  __shared__ _Float16 lK[64 * LDK];    // [kpos][hd]
  __shared__ _Float16 lV[64 * LDK];    // transposed: [hd][kpos]
  __shared__ _Float16 lP[8][16 * LDK]; // per-wave P scratch [qrow][kpos]

  const int tid  = threadIdx.x;
  const int lane = tid & 31;
  const int wave = tid >> 5;
  const int lr   = lane & 15;
  const int hl   = lane >> 4;

  const int bh   = blockIdx.y;
  const int b    = bh / Hq;
  const int h    = bh % Hq;
  const int kvbh = b * KVHq + (h >> 2);          // REP = 4
  const int qblock = blockIdx.x * 128;
  const int q0w    = qblock + wave * 16;

  HFrag qa[2];                                    // Q A-frags, 2 chunks over HD=64
  {
    const _Float16* qrow = qh + ((size_t)bh * Sq + (q0w + lr)) * HDq;
    #pragma unroll
    for (int c = 0; c < 2; ++c) {
      qa[c].q[0] = *(const uint4*)&qrow[c * 32 + hl * 8];
      qa[c].q[1] = *(const uint4*)&qrow[c * 32 + 16 + hl * 8];
      // fold softmax scale (HD^-0.5 = 0.125, power of two -> exact in f16)
      qa[c].v *= (_Float16)0.125f;
    }
  }

  v8f oacc[4];
  #pragma unroll
  for (int nf = 0; nf < 4; ++nf) oacc[nf] = {};
  float m_i[8], l_i[8];
  #pragma unroll
  for (int r = 0; r < 8; ++r) { m_i[r] = NEG_BIG; l_i[r] = 0.0f; }

  const int ntiles = blockIdx.x * 2 + 2;          // causal: kv tiles with k0 < qblock+128

  for (int t = 0; t < ntiles; ++t) {
    const int k0 = t * 64;
    __syncthreads();
    {
      // K tile: async memory->LDS copy (ASYNCcnt path, no VGPR transit).
      int row = tid >> 2, seg = tid & 3;
      unsigned lds_k = (unsigned)(uintptr_t)&lK[row * LDK + seg * 8];
      const _Float16* gk = &kh[((size_t)kvbh * Sq + k0 + row) * HDq + seg * 8];
      asm volatile("global_load_async_to_lds_b128 %0, %1, off"
                   :: "v"(lds_k), "v"(gk) : "memory");

      // V transposed: each thread covers 2 adjacent kpos rows at one hd-slice,
      // packing pairs (hd, kp),(hd, kp+1) into b32 LDS stores.
      int kp  = (tid >> 3) * 2;
      int hd0 = (tid & 7) * 8;
      H8 v0, v1;
      v0.q = *(const uint4*)&vh[((size_t)kvbh * Sq + k0 + kp) * HDq + hd0];
      v1.q = *(const uint4*)&vh[((size_t)kvbh * Sq + k0 + kp + 1) * HDq + hd0];
      #pragma unroll
      for (int j = 0; j < 8; ++j) {
        union { _Float16 h[2]; unsigned u; } pk;
        pk.h[0] = v0.h[j];
        pk.h[1] = v1.h[j];
        *(unsigned*)&lV[(hd0 + j) * LDK + kp] = pk.u;
      }
      asm volatile("s_wait_asynccnt 0x0" ::: "memory");
    }
    __syncthreads();

    // S = (Q*scale) K^T (16x64): batch 4 B-frags per K-chunk, then 4 WMMAs
    v8f sfr[4];
    #pragma unroll
    for (int nf = 0; nf < 4; ++nf) sfr[nf] = {};
    #pragma unroll
    for (int c = 0; c < 2; ++c) {
      HFrag kb[4];
      #pragma unroll
      for (int nf = 0; nf < 4; ++nf) {
        int row = nf * 16 + lr;
        kb[nf].q[0] = *(const uint4*)&lK[row * LDK + c * 32 + hl * 8];
        kb[nf].q[1] = *(const uint4*)&lK[row * LDK + c * 32 + 16 + hl * 8];
      }
      #pragma unroll
      for (int nf = 0; nf < 4; ++nf)
        sfr[nf] = wmma16(qa[c].v, kb[nf].v, sfr[nf]);
    }

    float nm[8];                                  // causal mask + online max
    #pragma unroll
    for (int r = 0; r < 8; ++r) nm[r] = m_i[r];
    #pragma unroll
    for (int nf = 0; nf < 4; ++nf) {
      int kj = k0 + nf * 16 + lr;
      #pragma unroll
      for (int r = 0; r < 8; ++r) {
        int qi = q0w + r + 8 * hl;
        float s = sfr[nf][r];
        if (kj > qi) s = NEG_BIG;
        sfr[nf][r] = s;
        nm[r] = fmaxf(nm[r], s);
      }
    }
    #pragma unroll
    for (int r = 0; r < 8; ++r)
      #pragma unroll
      for (int msk = 8; msk >= 1; msk >>= 1)
        nm[r] = fmaxf(nm[r], __shfl_xor(nm[r], msk, 32));

    float corr[8], rs[8];
    #pragma unroll
    for (int r = 0; r < 8; ++r) {
      corr[r] = __expf(m_i[r] - nm[r]);
      m_i[r]  = nm[r];
      rs[r]   = 0.0f;
    }
    #pragma unroll
    for (int nf = 0; nf < 4; ++nf) {
      #pragma unroll
      for (int r = 0; r < 8; ++r) {
        float p = __expf(sfr[nf][r] - m_i[r]);
        rs[r] += p;
        lP[wave][(r + 8 * hl) * LDK + nf * 16 + lr] = (_Float16)p;
      }
    }
    #pragma unroll
    for (int r = 0; r < 8; ++r) {
      #pragma unroll
      for (int msk = 8; msk >= 1; msk >>= 1)
        rs[r] += __shfl_xor(rs[r], msk, 32);
      l_i[r] = l_i[r] * corr[r] + rs[r];
    }
    #pragma unroll
    for (int nf = 0; nf < 4; ++nf)
      #pragma unroll
      for (int r = 0; r < 8; ++r) oacc[nf][r] *= corr[r];

    asm volatile("s_wait_dscnt 0" ::: "memory");  // P stores visible to own-wave reads

    HFrag pa[2];                                  // P as A-frags (K = kpos)
    #pragma unroll
    for (int c = 0; c < 2; ++c) {
      pa[c].q[0] = *(const uint4*)&lP[wave][lr * LDK + c * 32 + hl * 8];
      pa[c].q[1] = *(const uint4*)&lP[wave][lr * LDK + c * 32 + 16 + hl * 8];
    }
    #pragma unroll
    for (int c = 0; c < 2; ++c) {
      HFrag vb[4];
      #pragma unroll
      for (int nf = 0; nf < 4; ++nf) {
        int row = nf * 16 + lr;                   // lV row = hd column of V
        vb[nf].q[0] = *(const uint4*)&lV[row * LDK + c * 32 + hl * 8];
        vb[nf].q[1] = *(const uint4*)&lV[row * LDK + c * 32 + 16 + hl * 8];
      }
      #pragma unroll
      for (int nf = 0; nf < 4; ++nf)
        oacc[nf] = wmma16(pa[c].v, vb[nf].v, oacc[nf]);
    }
  }

  #pragma unroll
  for (int nf = 0; nf < 4; ++nf) {
    #pragma unroll
    for (int r = 0; r < 8; ++r) {
      int qi = q0w + r + 8 * hl;
      int hd = nf * 16 + lr;
      ctxh[((size_t)b * Sq + qi) * Eq + h * HDq + hd] =
          (_Float16)(oacc[nf][r] / l_i[r]);
    }
  }
}

extern "C" void kernel_launch(void* const* d_in, const int* in_sizes, int n_in,
                              void* d_out, int out_size, void* d_ws, size_t ws_size,
                              hipStream_t stream) {
  (void)in_sizes; (void)n_in; (void)out_size; (void)ws_size;
  const float* x    = (const float*)d_in[0];
  // d_in[1] = mask (implied by causal structure; unused)
  const float* wq_w = (const float*)d_in[2];
  const float* wq_b = (const float*)d_in[3];
  const float* wk_w = (const float*)d_in[4];
  const float* wk_b = (const float*)d_in[5];
  const float* wv_w = (const float*)d_in[6];
  const float* wv_b = (const float*)d_in[7];
  const float* wo_w = (const float*)d_in[8];
  const float* wo_b = (const float*)d_in[9];

  char* ws = (char*)d_ws;
  size_t off = 0;
  auto take = [&](size_t n_half) -> _Float16* {
    _Float16* p = (_Float16*)(ws + off);
    off += ((n_half * sizeof(_Float16)) + 255) & ~(size_t)255;
    return p;
  };
  const size_t M = (size_t)Bq * Sq;                   // 4096
  _Float16* xh  = take(M * Eq);                       // x in f16
  _Float16* wqh = take((size_t)Eq * Eq);
  _Float16* wkh = take((size_t)512 * Eq);
  _Float16* wvh = take((size_t)512 * Eq);
  _Float16* woh = take((size_t)Eq * Eq);
  _Float16* qh  = take((size_t)Bq * Hq * Sq * HDq);   // (B*H,S,HD)
  _Float16* kh  = take((size_t)Bq * KVHq * Sq * HDq); // (B*KVH,S,HD)
  _Float16* vh  = take((size_t)Bq * KVHq * Sq * HDq);
  _Float16* ch  = take(M * Eq);                       // ctx f16 (B,S,E)

  auto cvt = [&](const float* src, _Float16* dst, int n) {
    cvt_f32_f16<<<n / 256, 256, 0, stream>>>(src, dst, n);
  };
  cvt(x,    xh,  (int)(M * Eq));
  cvt(wq_w, wqh, Eq * Eq);
  cvt(wk_w, wkh, 512 * Eq);
  cvt(wv_w, wvh, 512 * Eq);
  cvt(wo_w, woh, Eq * Eq);

  gemm_f16_wmma<1><<<dim3(Eq / 128,  (int)(M / 128)), 256, 0, stream>>>(
      xh, wqh, wq_b, qh, (int)M, Eq, Eq, Hq);
  gemm_f16_wmma<1><<<dim3(512 / 128, (int)(M / 128)), 256, 0, stream>>>(
      xh, wkh, wk_b, kh, (int)M, 512, Eq, KVHq);
  gemm_f16_wmma<1><<<dim3(512 / 128, (int)(M / 128)), 256, 0, stream>>>(
      xh, wvh, wv_b, vh, (int)M, 512, Eq, KVHq);

  attn_flash_wmma<<<dim3(Sq / 128, Bq * Hq), 256, 0, stream>>>(qh, kh, vh, ch);

  gemm_f16_wmma<0><<<dim3(Eq / 128, (int)(M / 128)), 256, 0, stream>>>(
      ch, woh, wo_b, d_out, (int)M, Eq, Eq, 0);
}
